// StructuredStateRecurrence_10548439679752
// MI455X (gfx1250) — matrix operand order repaired and checked by reading, not compiled, not run on hardware
//
#include <hip/hip_runtime.h>
#include <hip/hip_bf16.h>
#include <cstdint>

typedef __attribute__((ext_vector_type(16))) __bf16 v16bf;
typedef __attribute__((ext_vector_type(8)))  float  v8f;

#define B_ 16
#define S_ 2048
#define D_ 1024
#define M_ 256

#define EPI_NONE  0
#define EPI_SILU  1
#define EPI_RESID 2

// round-to-nearest-even fp32 -> bf16 (raw u16)
__device__ __forceinline__ unsigned short f2bf(float f) {
  unsigned int u = __float_as_uint(f);
  u += 0x7FFFu + ((u >> 16) & 1u);
  return (unsigned short)(u >> 16);
}

// fast sigmoid: single v_exp + v_rcp (no IEEE division expansion)
__device__ __forceinline__ float fast_sigmoid(float x) {
  return __builtin_amdgcn_rcpf(1.0f + __expf(-x));
}

union __align__(16) Frag16 {
  v16bf v;
  unsigned short us[16];
  uint4 q[2];
};

// ---------------------------------------------------------------------------
// x (fp32, contiguous) -> bf16, vectorized 8 elements/thread-iter
// ---------------------------------------------------------------------------
__global__ void convert_x_kernel(const float* __restrict__ src,
                                 unsigned short* __restrict__ dst, long n8) {
  for (long i = blockIdx.x * (long)blockDim.x + threadIdx.x; i < n8;
       i += (long)gridDim.x * blockDim.x) {
    const float4* p = (const float4*)(src + i * 8);
    float4 a = p[0], b = p[1];
    union { uint4 q; unsigned short us[8]; } o;
    o.us[0] = f2bf(a.x); o.us[1] = f2bf(a.y); o.us[2] = f2bf(a.z); o.us[3] = f2bf(a.w);
    o.us[4] = f2bf(b.x); o.us[5] = f2bf(b.y); o.us[6] = f2bf(b.z); o.us[7] = f2bf(b.w);
    ((uint4*)dst)[i] = o.q;
  }
}

// ---------------------------------------------------------------------------
// Weight slice conversion: fp32 (row stride srcStride, starting col colOfs)
// -> bf16 row-major rows x cols
// ---------------------------------------------------------------------------
__global__ void convert_bf16_kernel(const float* __restrict__ src, int srcStride, int colOfs,
                                    unsigned short* __restrict__ dst, int rows, int cols) {
  int total = rows * cols;
  for (int i = blockIdx.x * blockDim.x + threadIdx.x; i < total; i += gridDim.x * blockDim.x) {
    int r = i / cols, c = i - r * cols;
    dst[i] = f2bf(src[(size_t)r * srcStride + colOfs + c]);
  }
}

// ---------------------------------------------------------------------------
// Projection GEMM: out[row, n] = epi( sum_k A[row,k]*W[n,k] + bias[n] )
// A: 32768 x 1024 bf16 row-major (pre-converted x)
// W: Ncols x 1024 bf16 row-major
// Wave tile: 16 rows x 64 cols (4 accumulators sharing one A fragment).
// Block = 8 waves = 2 row-tiles x 4 col-groups -> 32 rows x 256 cols.
// ---------------------------------------------------------------------------
__global__ __launch_bounds__(256) void gemm_proj_kernel(
    const unsigned short* __restrict__ Abf,
    const unsigned short* __restrict__ W,
    const float* __restrict__ bias,
    const float* __restrict__ resid,   // nullable; stride = Ncols
    float* __restrict__ out,
    int Ncols, int epi) {
  const int lane   = threadIdx.x & 31;
  const int wave   = threadIdx.x >> 5;
  const int m      = lane & 15;
  const int hiHalf = lane >> 4;                  // 0: lanes 0-15, 1: lanes 16-31
  const int row0   = blockIdx.x * 32 + (wave >> 2) * 16;
  const int n0     = blockIdx.y * 256 + (wave & 3) * 64;

  const unsigned short* arow = Abf + (size_t)(row0 + m) * (size_t)D_;
  const int kA = hiHalf * 16;      // A fragment K base per half-wave
  const int kB = hiHalf * 16;      // B fragment K base per half-wave

  v8f acc[4] = {{}, {}, {}, {}};

  for (int k0 = 0; k0 < D_; k0 += 32) {
    Frag16 a;
    a.q[0] = *(const uint4*)(arow + k0 + kA);        // K = kA+0..7
    a.q[1] = *(const uint4*)(arow + k0 + kA + 8);    // K = kA+8..15
#pragma unroll
    for (int j = 0; j < 4; ++j) {
      Frag16 b;
      const uint4* pb = (const uint4*)(W + (size_t)(n0 + j * 16 + m) * (size_t)D_ + k0 + kB);
      b.q[0] = pb[0]; b.q[1] = pb[1];
      acc[j] = __builtin_amdgcn_wmma_f32_16x16x32_bf16(false, a.v, false, b.v,
                                                       (short)0, acc[j], false, false);
    }
  }

#pragma unroll
  for (int j = 0; j < 4; ++j) {
#pragma unroll
    for (int r = 0; r < 8; ++r) {
      int row = row0 + r + hiHalf * 8;
      int col = n0 + j * 16 + m;
      float v = acc[j][r] + bias[col];
      if (epi == EPI_SILU)  v = v * __builtin_amdgcn_rcpf(1.0f + __expf(-v));
      if (epi == EPI_RESID) v += resid[(size_t)row * Ncols + col];
      out[(size_t)row * Ncols + col] = v;
    }
  }
}

// ---------------------------------------------------------------------------
// Sequential scan: one persistent workgroup, 32 waves (wave32), B=16 == WMMA M.
// Per step:
//   waves  0..15: f_pre tile (16 batches x 16 cols) = h_bf16 @ WfH^T (+xf) -> sigmoid
//   waves 16..31: u_pre tile                         = h_bf16 @ WuH^T (+xu) -> sigmoid
//   all:          h_new = f*h + u*xm_t  (fp32 in LDS, bf16 mirror for WMMA A)
//   all:          out_t = xo_t + h_new @ WoH^T  (2 col-tiles of 16 per wave)
// ---------------------------------------------------------------------------
__global__ __launch_bounds__(1024) void recurrence_kernel(
    const float* __restrict__ xm, const float* __restrict__ xf, const float* __restrict__ xu,
    const float* __restrict__ xo,
    const unsigned short* __restrict__ WfH,   // 256 x 256 bf16
    const unsigned short* __restrict__ WuH,   // 256 x 256 bf16
    const unsigned short* __restrict__ WoH,   // 1024 x 256 bf16
    float* __restrict__ out) {
  __shared__ __align__(16) float          h_lds[B_ * M_];
  __shared__ __align__(16) unsigned short hbf  [B_ * M_];
  __shared__ __align__(16) float          f_lds[B_ * M_];
  __shared__ __align__(16) float          u_lds[B_ * M_];

  const int tid    = threadIdx.x;
  const int lane   = tid & 31;
  const int wave   = tid >> 5;
  const int m      = lane & 15;
  const int hiHalf = lane >> 4;
  const int kA     = hiHalf * 16;
  const int kB     = hiHalf * 16;

  for (int i = tid; i < B_ * M_; i += 1024) { h_lds[i] = 0.0f; hbf[i] = 0; }
  __syncthreads();

  const bool isF = (wave < 16);
  const int  gN0 = (wave & 15) * 16;
  const unsigned short* Wg  = isF ? WfH : WuH;
  const float*          xg  = isF ? xf  : xu;
  float*                gds = isF ? f_lds : u_lds;

  for (int t = 0; t < S_; ++t) {
    // ---- gate preactivations via WMMA ----
    v8f acc = {};
#pragma unroll
    for (int k0 = 0; k0 < M_; k0 += 32) {
      Frag16 a, b;
      a.q[0] = *(const uint4*)(&hbf[m * M_ + k0 + kA]);
      a.q[1] = *(const uint4*)(&hbf[m * M_ + k0 + kA + 8]);
      const uint4* pb = (const uint4*)(Wg + (size_t)(gN0 + m) * M_ + k0 + kB);
      b.q[0] = pb[0]; b.q[1] = pb[1];
      acc = __builtin_amdgcn_wmma_f32_16x16x32_bf16(false, a.v, false, b.v,
                                                    (short)0, acc, false, false);
    }
#pragma unroll
    for (int r = 0; r < 8; ++r) {
      int bb = r + hiHalf * 8;
      int n  = gN0 + m;
      float pre = acc[r] + xg[((size_t)bb * S_ + t) * M_ + n];
      gds[bb * M_ + n] = fast_sigmoid(pre);
    }
    __syncthreads();

    // ---- h update: h = f*h + u*xm_t ----
    for (int i = tid; i < B_ * M_; i += 1024) {
      int bb = i >> 8, c = i & (M_ - 1);
      float hn = f_lds[i] * h_lds[i] + u_lds[i] * xm[((size_t)bb * S_ + t) * M_ + c];
      h_lds[i] = hn;
      hbf[i]   = f2bf(hn);
    }
    __syncthreads();

    // ---- out_t = xo_t + h_new @ WoH^T : 64 col-tiles, 2 per wave ----
#pragma unroll
    for (int tt = 0; tt < 2; ++tt) {
      int o0 = (wave * 2 + tt) * 16;
      v8f oacc = {};
#pragma unroll
      for (int k0 = 0; k0 < M_; k0 += 32) {
        Frag16 a, b;
        a.q[0] = *(const uint4*)(&hbf[m * M_ + k0 + kA]);
        a.q[1] = *(const uint4*)(&hbf[m * M_ + k0 + kA + 8]);
        const uint4* pb = (const uint4*)(WoH + (size_t)(o0 + m) * M_ + k0 + kB);
        b.q[0] = pb[0]; b.q[1] = pb[1];
        oacc = __builtin_amdgcn_wmma_f32_16x16x32_bf16(false, a.v, false, b.v,
                                                       (short)0, oacc, false, false);
      }
#pragma unroll
      for (int r = 0; r < 8; ++r) {
        int bb = r + hiHalf * 8;
        size_t idx = ((size_t)bb * S_ + t) * D_ + o0 + m;
        out[idx] = xo[idx] + oacc[r];
      }
    }
    __syncthreads();
  }

  // h_final (B x M), appended after the (B,S,D) outputs
  for (int i = tid; i < B_ * M_; i += 1024) {
    out[(size_t)B_ * S_ * D_ + i] = h_lds[i];
  }
}

// ---------------------------------------------------------------------------
extern "C" void kernel_launch(void* const* d_in, const int* in_sizes, int n_in,
                              void* d_out, int out_size, void* d_ws, size_t ws_size,
                              hipStream_t stream) {
  (void)in_sizes; (void)n_in; (void)out_size; (void)ws_size;
  const float* x  = (const float*)d_in[0];
  const float* Wi = (const float*)d_in[1];
  const float* bi = (const float*)d_in[2];
  const float* Wf = (const float*)d_in[3];
  const float* bf = (const float*)d_in[4];
  const float* Wu = (const float*)d_in[5];
  const float* bu = (const float*)d_in[6];
  const float* Wo = (const float*)d_in[7];
  const float* bo = (const float*)d_in[8];
  float* out = (float*)d_out;

  const size_t R = (size_t)B_ * S_;   // 32768 token rows
  char* ws = (char*)d_ws;
  size_t off = 0;
  auto carve = [&](size_t bytes) -> void* {
    void* p = ws + off;
    off += (bytes + 255) & ~(size_t)255;
    return p;
  };
  float* xm_ws = (float*)carve(R * M_ * sizeof(float));
  float* xf_ws = (float*)carve(R * M_ * sizeof(float));
  float* xu_ws = (float*)carve(R * M_ * sizeof(float));
  float* xo_ws = (float*)carve(R * D_ * sizeof(float));
  unsigned short* x_bf   = (unsigned short*)carve(R * (size_t)D_ * 2);
  unsigned short* Wi_bf  = (unsigned short*)carve((size_t)M_ * D_ * 2);
  unsigned short* WfHead = (unsigned short*)carve((size_t)M_ * D_ * 2);
  unsigned short* WfH_bf = (unsigned short*)carve((size_t)M_ * M_ * 2);
  unsigned short* WuHead = (unsigned short*)carve((size_t)M_ * D_ * 2);
  unsigned short* WuH_bf = (unsigned short*)carve((size_t)M_ * M_ * 2);
  unsigned short* WoHead = (unsigned short*)carve((size_t)D_ * D_ * 2);
  unsigned short* WoH_bf = (unsigned short*)carve((size_t)D_ * M_ * 2);

  // x -> bf16 (once; reused by all four projection GEMMs)
  convert_x_kernel<<<4096, 256, 0, stream>>>(x, x_bf, (long)(R * (size_t)D_ / 8));

  // weight slices -> bf16
  convert_bf16_kernel<<<512, 256, 0, stream>>>(Wi, D_,      0,  Wi_bf,  M_, D_);
  convert_bf16_kernel<<<512, 256, 0, stream>>>(Wf, D_ + M_, 0,  WfHead, M_, D_);
  convert_bf16_kernel<<<128, 256, 0, stream>>>(Wf, D_ + M_, D_, WfH_bf, M_, M_);
  convert_bf16_kernel<<<512, 256, 0, stream>>>(Wu, D_ + M_, 0,  WuHead, M_, D_);
  convert_bf16_kernel<<<128, 256, 0, stream>>>(Wu, D_ + M_, D_, WuH_bf, M_, M_);
  convert_bf16_kernel<<<512, 256, 0, stream>>>(Wo, D_ + M_, 0,  WoHead, D_, D_);
  convert_bf16_kernel<<<512, 256, 0, stream>>>(Wo, D_ + M_, D_, WoH_bf, D_, M_);

  // projections (block covers 32 rows x 256 cols)
  gemm_proj_kernel<<<dim3(R / 32, 1), 256, 0, stream>>>(x_bf, Wi_bf,  bi, nullptr, xm_ws, M_, EPI_SILU);
  gemm_proj_kernel<<<dim3(R / 32, 1), 256, 0, stream>>>(x_bf, WfHead, bf, nullptr, xf_ws, M_, EPI_NONE);
  gemm_proj_kernel<<<dim3(R / 32, 1), 256, 0, stream>>>(x_bf, WuHead, bu, nullptr, xu_ws, M_, EPI_NONE);
  gemm_proj_kernel<<<dim3(R / 32, 4), 256, 0, stream>>>(x_bf, WoHead, bo, x,       xo_ws, D_, EPI_RESID);

  // sequential scan: one persistent workgroup, 32 waves
  recurrence_kernel<<<1, 1024, 0, stream>>>(xm_ws, xf_ws, xu_ws, xo_ws,
                                            WfH_bf, WuH_bf, WoH_bf, out);
}
